// Classifier_52776558133272
// MI455X (gfx1250) — compile-verified, matrix operand order checked by reading, not run on hardware
//
#include <hip/hip_runtime.h>
#include <hip/hip_bf16.h>
#include <math.h>

typedef __attribute__((ext_vector_type(16))) _Float16 v16h;
typedef __attribute__((ext_vector_type(8)))  _Float16 v8h;
typedef __attribute__((ext_vector_type(2)))  _Float16 h2;
typedef __attribute__((ext_vector_type(8)))  float    v8f;

#define EPS 1e-5f

static inline int cdiv(int a, int b) { return (a + b - 1) / b; }

// ---------------------------------------------------------------------------
// WMMA GEMM:  C[M,N] = act( A[M,K] * B[N,K]^T + bias[N] )
// Block = 128 threads = 4 waves; block tile 64x64; each wave owns a 32x32
// quadrant = 2x2 WMMA tiles -> 4 v_wmma per K-step with fragment reuse.
// LDS tiles are f16 with 40-element (80 B) row stride so every fragment run
// is 16B-aligned and fetched as ds_load_b128.
// ---------------------------------------------------------------------------
__global__ __launch_bounds__(128)
void wmma_gemm_kernel(const float* __restrict__ A, const float* __restrict__ B,
                      const float* __restrict__ bias, float* __restrict__ C,
                      int M, int N, int K, int act) {
    __shared__ __align__(16) _Float16 As[64][40];
    __shared__ __align__(16) _Float16 Bs[64][40];
    const int tid  = threadIdx.x;
    const int wave = tid >> 5;
    const int lane = tid & 31;
    const int wm = wave >> 1, wn = wave & 1;     // 2x2 wave grid
    const int rowBase = blockIdx.y * 64;
    const int colBase = blockIdx.x * 64;
    const int lh  = lane >> 4;                   // lane half
    const int l16 = lane & 15;

    const bool dimsFull = (rowBase + 64 <= M) && (colBase + 64 <= N);

    v8f acc00 = {}, acc01 = {}, acc10 = {}, acc11 = {};

    // staging map: thread -> row r (0..63), col block c0 (0 or 16)
    const int sr = tid >> 1;
    const int sc = (tid & 1) * 16;

    for (int k0 = 0; k0 < K; k0 += 32) {
        if (dimsFull && (k0 + 32 <= K)) {
            // -------- interior fast path: no per-lane predication --------
            const float* ap = A + (size_t)(rowBase + sr) * K + k0 + sc;
            const float* bp = B + (size_t)(colBase + sr) * K + k0 + sc;
#pragma unroll
            for (int j = 0; j < 8; ++j) {
                h2 pa; pa[0] = (_Float16)ap[2 * j]; pa[1] = (_Float16)ap[2 * j + 1];
                *(h2*)&As[sr][sc + 2 * j] = pa;
                h2 pb; pb[0] = (_Float16)bp[2 * j]; pb[1] = (_Float16)bp[2 * j + 1];
                *(h2*)&Bs[sr][sc + 2 * j] = pb;
            }
        } else {
            // -------- boundary path: guarded, zero-padded --------
            const int gr = rowBase + sr;
            const int gn = colBase + sr;
#pragma unroll
            for (int j = 0; j < 8; ++j) {
                int gc = k0 + sc + 2 * j;
                h2 pa, pb;
                pa[0] = (_Float16)((gr < M && gc < K)     ? A[(size_t)gr * K + gc]     : 0.0f);
                pa[1] = (_Float16)((gr < M && gc + 1 < K) ? A[(size_t)gr * K + gc + 1] : 0.0f);
                pb[0] = (_Float16)((gn < N && gc < K)     ? B[(size_t)gn * K + gc]     : 0.0f);
                pb[1] = (_Float16)((gn < N && gc + 1 < K) ? B[(size_t)gn * K + gc + 1] : 0.0f);
                *(h2*)&As[sr][sc + 2 * j] = pa;
                *(h2*)&Bs[sr][sc + 2 * j] = pb;
            }
        }
        __syncthreads();

        // Fragment fetch (ds_load_b128 pairs).  16-bit A 16x32 layout:
        // lanes 0-15 hold K=0..7 & 16..23; lanes 16-31 hold K=8..15 & 24..31.
        const _Float16* a0p = &As[wm * 32 + l16][0];
        const _Float16* a1p = &As[wm * 32 + 16 + l16][0];
        const _Float16* b0p = &Bs[wn * 32 + l16][0];
        const _Float16* b1p = &Bs[wn * 32 + 16 + l16][0];
        v8h a0lo = *(const v8h*)(a0p + lh * 8);
        v8h a0hi = *(const v8h*)(a0p + 16 + lh * 8);
        v8h a1lo = *(const v8h*)(a1p + lh * 8);
        v8h a1hi = *(const v8h*)(a1p + 16 + lh * 8);
        v8h b0lo = *(const v8h*)(b0p + lh * 8);
        v8h b0hi = *(const v8h*)(b0p + 16 + lh * 8);
        v8h b1lo = *(const v8h*)(b1p + lh * 8);
        v8h b1hi = *(const v8h*)(b1p + 16 + lh * 8);
        v16h a0 = __builtin_shufflevector(a0lo, a0hi, 0,1,2,3,4,5,6,7,8,9,10,11,12,13,14,15);
        v16h a1 = __builtin_shufflevector(a1lo, a1hi, 0,1,2,3,4,5,6,7,8,9,10,11,12,13,14,15);
        v16h b0 = __builtin_shufflevector(b0lo, b0hi, 0,1,2,3,4,5,6,7,8,9,10,11,12,13,14,15);
        v16h b1 = __builtin_shufflevector(b1lo, b1hi, 0,1,2,3,4,5,6,7,8,9,10,11,12,13,14,15);

        acc00 = __builtin_amdgcn_wmma_f32_16x16x32_f16(false, a0, false, b0, (short)0, acc00, false, false);
        acc01 = __builtin_amdgcn_wmma_f32_16x16x32_f16(false, a0, false, b1, (short)0, acc01, false, false);
        acc10 = __builtin_amdgcn_wmma_f32_16x16x32_f16(false, a1, false, b0, (short)0, acc10, false, false);
        acc11 = __builtin_amdgcn_wmma_f32_16x16x32_f16(false, a1, false, b1, (short)0, acc11, false, false);
        __syncthreads();
    }

    // C/D layout: VGPR r -> row r (lanes 0-15) / row r+8 (lanes 16-31).
    const int r0 = rowBase + wm * 32;
    const int c0 = colBase + wn * 32;
    const int cA = c0 + l16, cB = c0 + 16 + l16;
    if (dimsFull) {
#pragma unroll
        for (int r = 0; r < 8; ++r) {
            int rowA = r0 + r + 8 * lh;
            int rowB = r0 + 16 + r + 8 * lh;
            float v00 = acc00[r], v01 = acc01[r], v10 = acc10[r], v11 = acc11[r];
            if (bias) { v00 += bias[cA]; v01 += bias[cB]; v10 += bias[cA]; v11 += bias[cB]; }
            if (act) {
                v00 = fmaxf(v00, 0.f); v01 = fmaxf(v01, 0.f);
                v10 = fmaxf(v10, 0.f); v11 = fmaxf(v11, 0.f);
            }
            C[(size_t)rowA * N + cA] = v00;
            C[(size_t)rowA * N + cB] = v01;
            C[(size_t)rowB * N + cA] = v10;
            C[(size_t)rowB * N + cB] = v11;
        }
    } else {
#pragma unroll
        for (int r = 0; r < 8; ++r) {
            int rowA = r0 + r + 8 * lh;
            int rowB = r0 + 16 + r + 8 * lh;
            float vals[4] = { acc00[r], acc01[r], acc10[r], acc11[r] };
            int rows[4] = { rowA, rowA, rowB, rowB };
            int cols[4] = { cA, cB, cA, cB };
#pragma unroll
            for (int q = 0; q < 4; ++q) {
                if (rows[q] < M && cols[q] < N) {
                    float v = vals[q];
                    if (bias) v += bias[cols[q]];
                    if (act) v = fmaxf(v, 0.0f);
                    C[(size_t)rows[q] * N + cols[q]] = v;
                }
            }
        }
    }
}

// ---------------------------------------------------------------------------
// Direct 3D conv, kernel 3^3, pad 1, stride 1.  in [BT,Cin,D,H,W], OIDHW w.
// ---------------------------------------------------------------------------
__global__ void conv3d_k3(const float* __restrict__ in, const float* __restrict__ w,
                          const float* __restrict__ bias, float* __restrict__ out,
                          int BT, int Cin, int Cout, int D, int H, int W) {
    int idx = blockIdx.x * blockDim.x + threadIdx.x;
    int total = BT * Cout * D * H * W;
    if (idx >= total) return;
    int x = idx % W; int t = idx / W;
    int y = t % H; t /= H;
    int z = t % D; t /= D;
    int co = t % Cout; int n = t / Cout;
    float acc = bias[co];
    for (int ci = 0; ci < Cin; ++ci) {
        const float* ip = in + ((size_t)n * Cin + ci) * D * H * W;
        const float* wp = w + ((size_t)co * Cin + ci) * 27;
        for (int kd = 0; kd < 3; ++kd) {
            int zz = z + kd - 1; if (zz < 0 || zz >= D) continue;
            for (int kh = 0; kh < 3; ++kh) {
                int yy = y + kh - 1; if (yy < 0 || yy >= H) continue;
                for (int kw = 0; kw < 3; ++kw) {
                    int xx = x + kw - 1; if (xx < 0 || xx >= W) continue;
                    acc += ip[((size_t)zz * H + yy) * W + xx] * wp[kd * 9 + kh * 3 + kw];
                }
            }
        }
    }
    out[idx] = acc;
}

// ---------------------------------------------------------------------------
// Max pool 2x2x2 stride 2 with per-dim padding.
// ---------------------------------------------------------------------------
__global__ void maxpool3d(const float* __restrict__ in, float* __restrict__ out,
                          int BT, int C, int D, int H, int W,
                          int Do, int Ho, int Wo, int pd, int ph, int pw) {
    int idx = blockIdx.x * blockDim.x + threadIdx.x;
    int total = BT * C * Do * Ho * Wo;
    if (idx >= total) return;
    int ox = idx % Wo; int t = idx / Wo;
    int oy = t % Ho; t /= Ho;
    int oz = t % Do; t /= Do;
    int c = t % C; int n = t / C;
    const float* ip = in + ((size_t)n * C + c) * D * H * W;
    float m = -3.0e38f;
    for (int kz = 0; kz < 2; ++kz) {
        int z = oz * 2 - pd + kz; if (z < 0 || z >= D) continue;
        for (int ky = 0; ky < 2; ++ky) {
            int y = oy * 2 - ph + ky; if (y < 0 || y >= H) continue;
            for (int kx = 0; kx < 2; ++kx) {
                int x = ox * 2 - pw + kx; if (x < 0 || x >= W) continue;
                m = fmaxf(m, ip[((size_t)z * H + y) * W + x]);
            }
        }
    }
    out[idx] = m;
}

// ---------------------------------------------------------------------------
// BatchNorm stats / apply.
// ---------------------------------------------------------------------------
__global__ void bn3d_stats(const float* __restrict__ x, float* mean, float* rstd,
                           int BT, int C, int S) {
    __shared__ float s1[256], s2[256];
    int c = blockIdx.x;
    int count = BT * S;
    float a = 0.f, b = 0.f;
    for (int i = threadIdx.x; i < count; i += blockDim.x) {
        int n = i / S, s = i % S;
        float v = x[((size_t)n * C + c) * S + s];
        a += v; b += v * v;
    }
    s1[threadIdx.x] = a; s2[threadIdx.x] = b;
    __syncthreads();
    for (int s = 128; s > 0; s >>= 1) {
        if (threadIdx.x < s) { s1[threadIdx.x] += s1[threadIdx.x + s]; s2[threadIdx.x] += s2[threadIdx.x + s]; }
        __syncthreads();
    }
    if (threadIdx.x == 0) {
        float m = s1[0] / count;
        float var = s2[0] / count - m * m;
        mean[c] = m; rstd[c] = rsqrtf(var + EPS);
    }
}

__global__ void bn3d_apply(float* __restrict__ x, const float* mean, const float* rstd,
                           const float* g, const float* b, int BT, int C, int S, int relu) {
    int idx = blockIdx.x * blockDim.x + threadIdx.x;
    int total = BT * C * S;
    if (idx >= total) return;
    int c = (idx / S) % C;
    float v = (x[idx] - mean[c]) * rstd[c] * g[c] + b[c];
    if (relu) v = fmaxf(v, 0.0f);
    x[idx] = v;
}

__global__ void bn1d_stats(const float* __restrict__ x, float* mean, float* rstd,
                           int R, int F) {
    __shared__ float s1[256], s2[256];
    int f = blockIdx.x;
    float a = 0.f, b = 0.f;
    for (int r = threadIdx.x; r < R; r += blockDim.x) {
        float v = x[(size_t)r * F + f];
        a += v; b += v * v;
    }
    s1[threadIdx.x] = a; s2[threadIdx.x] = b;
    __syncthreads();
    for (int s = 128; s > 0; s >>= 1) {
        if (threadIdx.x < s) { s1[threadIdx.x] += s1[threadIdx.x + s]; s2[threadIdx.x] += s2[threadIdx.x + s]; }
        __syncthreads();
    }
    if (threadIdx.x == 0) {
        float m = s1[0] / R;
        float var = s2[0] / R - m * m;
        mean[f] = m; rstd[f] = rsqrtf(var + EPS);
    }
}

__global__ void bn1d_apply(float* __restrict__ x, const float* mean, const float* rstd,
                           const float* g, const float* b, int R, int F, int relu) {
    int idx = blockIdx.x * blockDim.x + threadIdx.x;
    int total = R * F;
    if (idx >= total) return;
    int f = idx % F;
    float v = (x[idx] - mean[f]) * rstd[f] * g[f] + b[f];
    if (relu) v = fmaxf(v, 0.0f);
    x[idx] = v;
}

// ---------------------------------------------------------------------------
// GRU step: gi precomputed [B*T, 3H]; computes gh = h@Whh^T + bhh and updates.
// ---------------------------------------------------------------------------
__global__ void gru_step(const float* __restrict__ gi, const float* __restrict__ whh,
                         const float* __restrict__ bhh, const float* __restrict__ h_in,
                         float* __restrict__ h_out, float* __restrict__ yseq,
                         int t, int B, int T, int Hd) {
    int idx = blockIdx.x * blockDim.x + threadIdx.x;
    if (idx >= B * Hd) return;
    int b = idx / Hd, j = idx % Hd;
    const float* hb = h_in + (size_t)b * Hd;
    float gr = bhh[j], gz = bhh[Hd + j], gn = bhh[2 * Hd + j];
    const float* wr = whh + (size_t)j * Hd;
    const float* wz = whh + (size_t)(Hd + j) * Hd;
    const float* wn = whh + (size_t)(2 * Hd + j) * Hd;
    for (int k = 0; k < Hd; ++k) {
        float hv = hb[k];
        gr += hv * wr[k]; gz += hv * wz[k]; gn += hv * wn[k];
    }
    const float* gib = gi + ((size_t)b * T + t) * 3 * Hd;
    float r = 1.0f / (1.0f + expf(-(gib[j] + gr)));
    float z = 1.0f / (1.0f + expf(-(gib[Hd + j] + gz)));
    float n = tanhf(gib[2 * Hd + j] + r * gn);
    float h = (1.0f - z) * n + z * hb[j];
    h_out[idx] = h;
    yseq[((size_t)b * T + t) * Hd + j] = h;
}

// ---------------------------------------------------------------------------
// Misc elementwise kernels.
// ---------------------------------------------------------------------------
__global__ void fill_kernel(float* p, float v, int n) {
    int i = blockIdx.x * blockDim.x + threadIdx.x;
    if (i < n) p[i] = v;
}

__global__ void build_feats(const float* __restrict__ xyz, const float* __restrict__ enc,
                            const int* __restrict__ batch, float* __restrict__ feats, int N) {
    int idx = blockIdx.x * blockDim.x + threadIdx.x;
    if (idx >= N * 515) return;
    int n = idx / 515, f = idx % 515;
    feats[idx] = (f < 3) ? xyz[(size_t)n * 3 + f] : enc[(size_t)batch[n] * 512 + (f - 3)];
}

__device__ inline void atomicMaxF(float* addr, float val) {
    int* ai = reinterpret_cast<int*>(addr);
    int old = __float_as_int(*addr);
    while (val > __int_as_float(old)) {
        int assumed = old;
        old = atomicCAS(ai, assumed, __float_as_int(val));
        if (old == assumed) break;
    }
}

// ---------------------------------------------------------------------------
// GATv2 edge-phase kernels.
// ---------------------------------------------------------------------------
__global__ void gat_edge_logits(const float* __restrict__ xl, const float* __restrict__ xr,
                                const float* __restrict__ att, const int* __restrict__ ei,
                                int E0, int E, int H, int C,
                                float* __restrict__ logits, float* __restrict__ lmax) {
    int idx = blockIdx.x * blockDim.x + threadIdx.x;
    if (idx >= E * H) return;
    int e = idx / H, h = idx % H;
    int src, dst;
    if (e < E0) { src = ei[e]; dst = ei[E0 + e]; }
    else        { src = dst = e - E0; }
    const float* pl = xl + ((size_t)src * H + h) * C;
    const float* pr = xr + ((size_t)dst * H + h) * C;
    const float* pa = att + (size_t)h * C;
    float s = 0.f;
    for (int c = 0; c < C; ++c) {
        float v = pl[c] + pr[c];
        v = v > 0.f ? v : 0.2f * v;           // leaky_relu 0.2
        s += v * pa[c];
    }
    logits[idx] = s;
    atomicMaxF(lmax + (size_t)dst * H + h, s);
}

__global__ void gat_edge_exp(const float* __restrict__ logits, const float* __restrict__ lmax,
                             const int* __restrict__ ei, int E0, int E, int H,
                             float* __restrict__ exb, float* __restrict__ den,
                             float* __restrict__ cnt) {
    int idx = blockIdx.x * blockDim.x + threadIdx.x;
    if (idx >= E * H) return;
    int e = idx / H, h = idx % H;
    int dst = (e < E0) ? ei[E0 + e] : (e - E0);
    float ex = expf(logits[idx] - lmax[(size_t)dst * H + h]);
    exb[idx] = ex;
    atomicAdd(den + (size_t)dst * H + h, ex);
    if (h == 0) atomicAdd(cnt + dst, 1.0f);
}

__global__ void gat_edge_scatter(const float* __restrict__ xl, const float* __restrict__ exb,
                                 const float* __restrict__ den, const int* __restrict__ ei,
                                 int E0, int E, int H, int C, float* __restrict__ agg) {
    int idx = blockIdx.x * blockDim.x + threadIdx.x;
    if (idx >= E * H) return;
    int e = idx / H, h = idx % H;
    int src, dst;
    if (e < E0) { src = ei[e]; dst = ei[E0 + e]; }
    else        { src = dst = e - E0; }
    float alpha = exb[idx] / den[(size_t)dst * H + h];
    const float* pl = xl + ((size_t)src * H + h) * C;
    float* po = agg + ((size_t)dst * H + h) * C;
    for (int c = 0; c < C; ++c)
        atomicAdd(po + c, alpha * pl[c]);
}

__global__ void gat_finalize(const float* __restrict__ agg, const float* __restrict__ cnt,
                             const float* __restrict__ bias, float* __restrict__ out,
                             int N, int HC) {
    int idx = blockIdx.x * blockDim.x + threadIdx.x;
    if (idx >= N * HC) return;
    int n = idx / HC, j = idx % HC;
    out[idx] = agg[idx] / fmaxf(cnt[n], 1.0f) + bias[j];
}

__global__ void softmax2(const float* __restrict__ in, float* __restrict__ out, int N) {
    int n = blockIdx.x * blockDim.x + threadIdx.x;
    if (n >= N) return;
    float a = in[(size_t)n * 2], b = in[(size_t)n * 2 + 1];
    float m = fmaxf(a, b);
    float ea = expf(a - m), eb = expf(b - m);
    float s = ea + eb;
    out[(size_t)n * 2]     = ea / s;
    out[(size_t)n * 2 + 1] = eb / s;
}

// ---------------------------------------------------------------------------
// Host orchestration.
// ---------------------------------------------------------------------------
extern "C" void kernel_launch(void* const* d_in, const int* in_sizes, int n_in,
                              void* d_out, int out_size, void* d_ws, size_t ws_size,
                              hipStream_t stream) {
    (void)in_sizes; (void)n_in; (void)out_size; (void)ws_size;

    const float* X      = (const float*)d_in[0];
    const float* xyz    = (const float*)d_in[1];
    const float* c1w = (const float*)d_in[2];  const float* c1b = (const float*)d_in[3];
    const float* b1g = (const float*)d_in[4];  const float* b1b = (const float*)d_in[5];
    const float* c2w = (const float*)d_in[6];  const float* c2b = (const float*)d_in[7];
    const float* b2g = (const float*)d_in[8];  const float* b2b = (const float*)d_in[9];
    const float* c3w = (const float*)d_in[10]; const float* c3b = (const float*)d_in[11];
    const float* b3g = (const float*)d_in[12]; const float* b3b = (const float*)d_in[13];
    const float* f1w = (const float*)d_in[14]; const float* f1b = (const float*)d_in[15];
    const float* b4g = (const float*)d_in[16]; const float* b4b = (const float*)d_in[17];
    const float* f2w = (const float*)d_in[18]; const float* f2b = (const float*)d_in[19];
    const float* b5g = (const float*)d_in[20]; const float* b5b = (const float*)d_in[21];
    const float* g0wih = (const float*)d_in[22]; const float* g0whh = (const float*)d_in[23];
    const float* g0bih = (const float*)d_in[24]; const float* g0bhh = (const float*)d_in[25];
    const float* g1wih = (const float*)d_in[26]; const float* g1whh = (const float*)d_in[27];
    const float* g1bih = (const float*)d_in[28]; const float* g1bhh = (const float*)d_in[29];
    const float* lw  = (const float*)d_in[30]; const float* lb  = (const float*)d_in[31];
    const float* n0g = (const float*)d_in[32]; const float* n0b = (const float*)d_in[33];
    const float* a1wl = (const float*)d_in[34]; const float* a1wr = (const float*)d_in[35];
    const float* a1at = (const float*)d_in[36]; const float* a1b  = (const float*)d_in[37];
    const float* n1g = (const float*)d_in[38]; const float* n1b = (const float*)d_in[39];
    const float* a2wl = (const float*)d_in[40]; const float* a2wr = (const float*)d_in[41];
    const float* a2at = (const float*)d_in[42]; const float* a2b  = (const float*)d_in[43];
    const float* n2g = (const float*)d_in[44]; const float* n2b = (const float*)d_in[45];
    const float* a3wl = (const float*)d_in[46]; const float* a3wr = (const float*)d_in[47];
    const float* a3at = (const float*)d_in[48]; const float* a3b  = (const float*)d_in[49];
    const float* n3g = (const float*)d_in[50]; const float* n3b = (const float*)d_in[51];
    const float* a456wl = (const float*)d_in[52]; const float* a456wr = (const float*)d_in[53];
    const float* a456at = (const float*)d_in[54]; const float* a456b  = (const float*)d_in[55];
    const float* n456g  = (const float*)d_in[56]; const float* n456b  = (const float*)d_in[57];
    const float* a7wl = (const float*)d_in[58]; const float* a7wr = (const float*)d_in[59];
    const float* a7at = (const float*)d_in[60]; const float* a7b  = (const float*)d_in[61];
    const int* eidx  = (const int*)d_in[62];
    const int* batch = (const int*)d_in[63];
    float* outp = (float*)d_out;

    const int BT = 200, NN = 32000, E0 = 512000, EE = E0 + NN;

    float* ws = (float*)d_ws;
    size_t off = 0;
    float* bn_mean = ws + off; off += 1024;
    float* bn_rstd = ws + off; off += 1024;
    float* h_a = ws + off; off += 2048;
    float* h_b = ws + off; off += 2048;
    float* encb = ws + off; off += 4096;
    size_t arena = off;

    // Phase A (CNN + GRU) arena
    size_t oa = arena;
    float* conv1o = ws + oa; oa += (size_t)BT * 64 * 180;
    float* conv2o = ws + oa; oa += (size_t)BT * 64 * 180;
    float* pool1  = ws + oa; oa += (size_t)BT * 64 * 48;
    float* conv3o = ws + oa; oa += (size_t)BT * 256 * 48;
    float* pool2  = ws + oa; oa += (size_t)BT * 2048;
    float* fc1o   = ws + oa; oa += (size_t)BT * 512;
    float* fc2o   = ws + oa; oa += (size_t)BT * 256;
    float* gib    = ws + oa; oa += (size_t)BT * 768;
    float* gru0o  = ws + oa; oa += (size_t)BT * 256;
    float* gru1o  = ws + oa; oa += (size_t)BT * 256;

    // Phase B (GAT) arena — overlays phase A (stream-ordered reuse)
    size_t ob = arena;
    float* feats  = ws + ob; ob += (size_t)NN * 515;
    float* hb0    = ws + ob; ob += (size_t)NN * 256;
    float* hb1    = ws + ob; ob += (size_t)NN * 256;
    float* xlb    = ws + ob; ob += (size_t)NN * 256;
    float* xrb    = ws + ob; ob += (size_t)NN * 256;
    float* aggb   = ws + ob; ob += (size_t)NN * 256;
    float* logb   = ws + ob; ob += (size_t)EE * 2;
    float* exb    = ws + ob; ob += (size_t)EE * 2;
    float* lmaxb  = ws + ob; ob += (size_t)NN * 2;
    float* denb   = ws + ob; ob += (size_t)NN * 2;
    float* cntb   = ws + ob; ob += (size_t)NN;

    auto gemm = [&](const float* A, const float* B, const float* bias, float* C,
                    int M, int N, int K, int act) {
        dim3 g(cdiv(N, 64), cdiv(M, 64));
        wmma_gemm_kernel<<<g, 128, 0, stream>>>(A, B, bias, C, M, N, K, act);
    };
    auto bn1 = [&](float* x, const float* g, const float* b, int R, int F, int relu) {
        bn1d_stats<<<F, 256, 0, stream>>>(x, bn_mean, bn_rstd, R, F);
        bn1d_apply<<<cdiv(R * F, 256), 256, 0, stream>>>(x, bn_mean, bn_rstd, g, b, R, F, relu);
    };
    auto bn3 = [&](float* x, const float* g, const float* b, int C, int S) {
        bn3d_stats<<<C, 256, 0, stream>>>(x, bn_mean, bn_rstd, BT, C, S);
        bn3d_apply<<<cdiv(BT * C * S, 256), 256, 0, stream>>>(x, bn_mean, bn_rstd, g, b, BT, C, S, 1);
    };
    auto fill = [&](float* p, float v, int n) {
        fill_kernel<<<cdiv(n, 256), 256, 0, stream>>>(p, v, n);
    };

    // ---- CNN ----
    conv3d_k3<<<cdiv(BT * 64 * 180, 256), 256, 0, stream>>>(X, c1w, c1b, conv1o, BT, 5, 64, 5, 6, 6);
    bn3(conv1o, b1g, b1b, 64, 180);
    conv3d_k3<<<cdiv(BT * 64 * 180, 256), 256, 0, stream>>>(conv1o, c2w, c2b, conv2o, BT, 64, 64, 5, 6, 6);
    bn3(conv2o, b2g, b2b, 64, 180);
    maxpool3d<<<cdiv(BT * 64 * 48, 256), 256, 0, stream>>>(conv2o, pool1, BT, 64, 5, 6, 6, 3, 4, 4, 1, 1, 1);
    conv3d_k3<<<cdiv(BT * 256 * 48, 256), 256, 0, stream>>>(pool1, c3w, c3b, conv3o, BT, 64, 256, 3, 4, 4);
    bn3(conv3o, b3g, b3b, 256, 48);
    maxpool3d<<<cdiv(BT * 2048, 256), 256, 0, stream>>>(conv3o, pool2, BT, 256, 3, 4, 4, 2, 2, 2, 1, 0, 0);

    // ---- FC head ----
    gemm(pool2, f1w, f1b, fc1o, BT, 512, 2048, 0); bn1(fc1o, b4g, b4b, BT, 512, 1);
    gemm(fc1o, f2w, f2b, fc2o, BT, 256, 512, 0);   bn1(fc2o, b5g, b5b, BT, 256, 1);

    // ---- GRU layer 0 (input projection hoisted into one WMMA GEMM) ----
    gemm(fc2o, g0wih, g0bih, gib, BT, 768, 256, 0);
    fill(h_a, 0.0f, 2048);
    {
        float* hin = h_a; float* hout = h_b;
        for (int t = 0; t < 25; ++t) {
            gru_step<<<8, 256, 0, stream>>>(gib, g0whh, g0bhh, hin, hout, gru0o, t, 8, 25, 256);
            float* tmp = hin; hin = hout; hout = tmp;
        }
    }
    // ---- GRU layer 1 ----
    gemm(gru0o, g1wih, g1bih, gib, BT, 768, 256, 0);
    fill(h_a, 0.0f, 2048);
    {
        float* hin = h_a; float* hout = h_b;
        for (int t = 0; t < 25; ++t) {
            gru_step<<<8, 256, 0, stream>>>(gib, g1whh, g1bhh, hin, hout, gru1o, t, 8, 25, 256);
            float* tmp = hin; hin = hout; hout = tmp;
        }
    }
    // ---- encoder: [8, 6400] @ lw^T, ReLU ----
    gemm(gru1o, lw, lb, encb, 8, 512, 6400, 1);

    // ---- node features + BN ----
    build_feats<<<cdiv(NN * 515, 256), 256, 0, stream>>>(xyz, encb, batch, feats, NN);
    bn1(feats, n0g, n0b, NN, 515, 0);

    // ---- GATv2 layers ----
    auto gat = [&](const float* hin, int Fin, int H, int C,
                   const float* wl, const float* wr, const float* att, const float* bias,
                   float* hout) {
        int HC = H * C;
        gemm(hin, wl, nullptr, xlb, NN, HC, Fin, 0);
        gemm(hin, wr, nullptr, xrb, NN, HC, Fin, 0);
        fill(lmaxb, -3.0e38f, NN * H);
        fill(denb, 0.0f, NN * H);
        fill(cntb, 0.0f, NN);
        fill(aggb, 0.0f, NN * HC);
        int EH = EE * H;
        gat_edge_logits<<<cdiv(EH, 256), 256, 0, stream>>>(xlb, xrb, att, eidx, E0, EE, H, C, logb, lmaxb);
        gat_edge_exp<<<cdiv(EH, 256), 256, 0, stream>>>(logb, lmaxb, eidx, E0, EE, H, exb, denb, cntb);
        gat_edge_scatter<<<cdiv(EH, 256), 256, 0, stream>>>(xlb, exb, denb, eidx, E0, EE, H, C, aggb);
        gat_finalize<<<cdiv(NN * HC, 256), 256, 0, stream>>>(aggb, cntb, bias, hout, NN, HC);
    };

    gat(feats, 515, 2, 128, a1wl, a1wr, a1at, a1b, hb0); bn1(hb0, n1g, n1b, NN, 256, 1);
    gat(hb0,   256, 2, 128, a2wl, a2wr, a2at, a2b, hb1); bn1(hb1, n2g, n2b, NN, 256, 1);
    gat(hb1,   256, 1, 128, a3wl, a3wr, a3at, a3b, hb0); bn1(hb0, n3g, n3b, NN, 128, 1);

    float* cur = hb0; float* nxt = hb1;
    for (int i = 0; i < 3; ++i) {
        gat(cur, 128, 1, 128, a456wl + (size_t)i * 128 * 128, a456wr + (size_t)i * 128 * 128,
            a456at + (size_t)i * 128, a456b + (size_t)i * 128, nxt);
        bn1(nxt, n456g + (size_t)i * 128, n456b + (size_t)i * 128, NN, 128, 1);
        float* tmp = cur; cur = nxt; nxt = tmp;
    }

    gat(cur, 128, 1, 2, a7wl, a7wr, a7at, a7b, nxt);
    softmax2<<<cdiv(NN, 256), 256, 0, stream>>>(nxt, outp, NN);
}
